// FlashAttention_26225070309487
// MI455X (gfx1250) — matrix operand order repaired and checked by reading, not compiled, not run on hardware
//
#include <hip/hip_runtime.h>
#include <hip/hip_bf16.h>

typedef __attribute__((ext_vector_type(16))) _Float16 v16h;
typedef __attribute__((ext_vector_type(8)))  float    v8f;

#define WMMA_F16(a, b, c) \
  __builtin_amdgcn_wmma_f32_16x16x32_f16(false, (a), false, (b), (short)0, (c), false, false)

static constexpr int Bn = 2, Nn = 2048, Cn = 1024, Hn = 16, Dn = 64;

#if defined(__has_builtin)
#if __has_builtin(__builtin_amdgcn_tensor_load_to_lds) && \
    __has_builtin(__builtin_amdgcn_s_wait_tensorcnt)
#define HAVE_TDM 1
#endif
#endif
#ifndef HAVE_TDM
#define HAVE_TDM 0
#endif

// ---------------------------------------------------------------------------
// Fragment loaders matching CDNA5 wave32 WMMA VGPR layouts (CDNA5 ISA).
// Both read two contiguous 16B runs per lane -> ds_load_b128 pairs.
// A 16x32 f16: lanes 0-15 row M=lane, K{0..7,16..23}; lanes 16-31 K{8..15,24..31}
// ---------------------------------------------------------------------------
__device__ inline v16h load_a_frag(const _Float16* base, int ld, int row0, int k0, int lane) {
  v16h a;
  const _Float16* p = base + (size_t)(row0 + (lane & 15)) * ld + k0 + ((lane >> 4) << 3);
#pragma unroll
  for (int i = 0; i < 4; ++i) {
    a[2 * i]     = p[2 * i];
    a[2 * i + 1] = p[2 * i + 1];
    a[2 * i + 8] = p[16 + 2 * i];
    a[2 * i + 9] = p[16 + 2 * i + 1];
  }
  return a;
}

// B 32x16 fragment where logical B[k][n] is stored transposed: buf[n*ld + k].
// lanes 0-15: col n = lane, K{0..7,16..23}; lanes 16-31: K{8..15,24..31} (matches A).
__device__ inline v16h load_bt_frag(const _Float16* base, int ld, int k0, int col0, int lane) {
  v16h b;
  const _Float16* p = base + (size_t)(col0 + (lane & 15)) * ld + k0 + ((lane >> 4) << 3);
#pragma unroll
  for (int i = 0; i < 16; ++i) b[i] = p[i < 8 ? i : i + 8];
  return b;
}

#if HAVE_TDM
typedef __attribute__((ext_vector_type(4))) unsigned int u32x4_t;
typedef __attribute__((ext_vector_type(8))) int          i32x8_t;
typedef __attribute__((ext_vector_type(4))) int          i32x4_t;

// TDM: load a 64x64 f16 row-major tile (row stride 64) into LDS with a
// 144-byte LDS row stride (128B data + 16B pad via pad_interval/pad_amount).
// D# bitfields per CDNA5 ISA.
// clang-23 toolchain: 6-arg builtin (g0 u32x4, g1 i32x8, g2 i32x4, g3 i32x4,
// extra i32x8, cpol imm).
__device__ inline void tdm_load_64x64_f16(const _Float16* gsrc, unsigned lds_off,
                                          unsigned tensor_rows) {
  unsigned long long ga = (unsigned long long)(uintptr_t)gsrc;
  u32x4_t g0;
  g0.x = 1u;                                                  // count=1 (valid), user mode
  g0.y = lds_off;                                             // lds_addr (bytes)
  g0.z = (unsigned)ga;                                        // global_addr[31:0]
  g0.w = (unsigned)((ga >> 32) & 0x01FFFFFFu) | 0x80000000u;  // global_addr[56:32], type=2
  i32x8_t g1;
  g1[0] = (1 << 16)        // data_size = 2 bytes
        | (1 << 20)        // pad_enable
        | (4 << 22)        // pad_interval: 2^(4+1)=32 DWORDs = 128B row
        | (3 << 25);       // pad_amount: 4 DWORDs = 16B  -> 144B LDS stride
  g1[1] = (int)(64u << 16);            // tensor_dim0 lo16 @ bits[63:48]
  g1[2] = (int)(tensor_rows << 16);    // tensor_dim0 hi16 | tensor_dim1 lo16
  g1[3] = (int)(64u << 16);            // tensor_dim1 hi16 | tile_dim0 = 64
  g1[4] = 64;                          // tile_dim1 = 64, tile_dim2 = 0
  g1[5] = 64;                          // tensor_dim0_stride = 64 elements
  g1[6] = 0;
  g1[7] = 0;
  i32x4_t z4;
  z4[0] = z4[1] = z4[2] = z4[3] = 0;   // 2D tensor: groups 2/3 unused
  i32x8_t z8;
#pragma unroll
  for (int i = 0; i < 8; ++i) z8[i] = 0;
  __builtin_amdgcn_tensor_load_to_lds(g0, g1, z4, z4, z8, 0);
}
#endif

// ---------------------------------------------------------------------------
// Generic WMMA GEMM: C[M,N] = A[M,K] * B[K,N] (+bias), f16 math / f32 accum.
// 128x128 block tile, 8 waves of 32x64, K-step 32. B tile stored transposed
// in LDS so all fragment reads are ds_load_b128.
// ---------------------------------------------------------------------------
template <bool A_HALF, bool OUT_HALF, bool BIAS>
__global__ __launch_bounds__(256) void gemm_wmma_kernel(
    const void* __restrict__ Ain, const float* __restrict__ Bf32,
    const float* __restrict__ bias, void* __restrict__ Cout,
    int M, int N, int K) {
  __shared__ __attribute__((aligned(16))) _Float16 As[128 * 40];  // [row][k]
  __shared__ __attribute__((aligned(16))) _Float16 Bs[128 * 40];  // [col][k] (transposed)

  const int t = threadIdx.x;
  const int lane = t & 31;
  const int wave = t >> 5;
  const int wr = wave & 3;   // 4 row-groups of 32
  const int wc = wave >> 2;  // 2 col-groups of 64
  const int tileM = blockIdx.y * 128;
  const int tileN = blockIdx.x * 128;

  const float*    Af = (const float*)Ain;
  const _Float16* Ah = (const _Float16*)Ain;

  v8f acc[2][4];
#pragma unroll
  for (int mt = 0; mt < 2; ++mt)
#pragma unroll
    for (int nt = 0; nt < 4; ++nt)
#pragma unroll
      for (int j = 0; j < 8; ++j) acc[mt][nt][j] = 0.0f;

  for (int k0 = 0; k0 < K; k0 += 32) {
    __syncthreads();
    // A tile 128x32: pack 2 halves per b32 LDS store
#pragma unroll
    for (int i = 0; i < 8; ++i) {
      int idx = t + i * 256;            // 2048 pairs
      int r = idx >> 4;
      int c2 = (idx & 15) << 1;
      unsigned u;
      if (A_HALF) {
        u = *(const unsigned*)&Ah[(size_t)(tileM + r) * K + k0 + c2];
      } else {
        float v0 = Af[(size_t)(tileM + r) * K + k0 + c2];
        float v1 = Af[(size_t)(tileM + r) * K + k0 + c2 + 1];
        union { _Float16 h[2]; unsigned uu; } pk;
        pk.h[0] = (_Float16)v0;
        pk.h[1] = (_Float16)v1;
        u = pk.uu;
      }
      *(unsigned*)&As[r * 40 + c2] = u;
    }
    // B tile 32x128 -> transposed [col][k], 2 k per thread (both rows coalesced)
#pragma unroll
    for (int i = 0; i < 8; ++i) {
      int idx = t + i * 256;            // 2048 pairs
      int c = idx & 127;
      int r2 = (idx >> 7) << 1;
      float v0 = Bf32[(size_t)(k0 + r2) * N + tileN + c];
      float v1 = Bf32[(size_t)(k0 + r2 + 1) * N + tileN + c];
      union { _Float16 h[2]; unsigned uu; } pk;
      pk.h[0] = (_Float16)v0;
      pk.h[1] = (_Float16)v1;
      *(unsigned*)&Bs[c * 40 + r2] = pk.uu;
    }
    __syncthreads();

    v16h afr[2], bfr[4];
#pragma unroll
    for (int mt = 0; mt < 2; ++mt) afr[mt] = load_a_frag(As, 40, wr * 32 + mt * 16, 0, lane);
#pragma unroll
    for (int nt = 0; nt < 4; ++nt) bfr[nt] = load_bt_frag(Bs, 40, 0, wc * 64 + nt * 16, lane);
#pragma unroll
    for (int mt = 0; mt < 2; ++mt)
#pragma unroll
      for (int nt = 0; nt < 4; ++nt) acc[mt][nt] = WMMA_F16(afr[mt], bfr[nt], acc[mt][nt]);
  }

#pragma unroll
  for (int mt = 0; mt < 2; ++mt)
#pragma unroll
    for (int nt = 0; nt < 4; ++nt) {
      int col = tileN + wc * 64 + nt * 16 + (lane & 15);
#pragma unroll
      for (int j = 0; j < 8; ++j) {
        int row = tileM + wr * 32 + mt * 16 + j + ((lane >> 4) << 3);
        float v = acc[mt][nt][j];
        if (BIAS) v += bias[col];
        if (OUT_HALF) ((_Float16*)Cout)[(size_t)row * N + col] = (_Float16)v;
        else          ((float*)Cout)[(size_t)row * N + col] = v;
      }
    }
}

// ---------------------------------------------------------------------------
// RoPE + repack qkv [B,N,3,H,d] -> q,k,v [B,H,N,d] f16. One thread per pair.
// ---------------------------------------------------------------------------
__global__ __launch_bounds__(256) void rope_pack_kernel(
    const _Float16* __restrict__ qkv, _Float16* __restrict__ qo,
    _Float16* __restrict__ ko, _Float16* __restrict__ vo) {
  int gid = blockIdx.x * 256 + threadIdx.x;  // B*H*N*32 = 2^21
  int i = gid & 31;
  int n = (gid >> 5) & (Nn - 1);
  int h = (gid >> 16) & (Hn - 1);
  int b = gid >> 20;

  size_t row = ((size_t)b * Nn + n) * (3 * Cn);
  int cq = h * Dn + i;
  float q1 = (float)qkv[row + cq],      q2 = (float)qkv[row + cq + 32];
  float k1 = (float)qkv[row + Cn + cq], k2 = (float)qkv[row + Cn + cq + 32];
  _Float16 v1 = qkv[row + 2 * Cn + cq], v2 = qkv[row + 2 * Cn + cq + 32];

  float ang = (float)n * __powf(10000.0f, -(float)i * (1.0f / 32.0f));
  float cs = __cosf(ang), sn = __sinf(ang);

  size_t o = (((size_t)b * Hn + h) * Nn + n) * Dn + i;
  qo[o]      = (_Float16)(q1 * cs - q2 * sn);
  qo[o + 32] = (_Float16)(q1 * sn + q2 * cs);
  ko[o]      = (_Float16)(k1 * cs - k2 * sn);
  ko[o + 32] = (_Float16)(k1 * sn + k2 * cs);
  vo[o]      = v1;
  vo[o + 32] = v2;
}

// ---------------------------------------------------------------------------
// Flash attention: one block per (b, h, 128 q-rows); 8 waves x 16 q-rows.
// K tile staged via TDM (tensor_load_to_lds, stride-72 rows synthesized with
// descriptor padding); V staged transposed [d][key] for b128 fragment reads.
// ---------------------------------------------------------------------------
__global__ __launch_bounds__(256) void flash_attn_kernel(
    const _Float16* __restrict__ qg, const _Float16* __restrict__ kg,
    const _Float16* __restrict__ vg, _Float16* __restrict__ og) {
  __shared__ __attribute__((aligned(16))) _Float16 PQ[8 * 16 * 72];  // Q stage, then P tiles
  __shared__ __attribute__((aligned(16))) _Float16 Ks[64 * 72];      // [key][d]
  __shared__ __attribute__((aligned(16))) _Float16 Vt[64 * 72];      // [d][key] transposed

  const int t = threadIdx.x, lane = t & 31, wave = t >> 5;
  const int qt = blockIdx.x & 15;
  const int h  = (blockIdx.x >> 4) & 15;
  const int b  = blockIdx.x >> 8;

  const size_t headbase = ((size_t)b * Hn + h) * Nn * Dn;

  {  // stage Q tile 128x64 (stride 72), 2 halves per b32
    const _Float16* qsrc = qg + headbase + (size_t)qt * 128 * Dn;
#pragma unroll
    for (int i = 0; i < 16; ++i) {
      int idx = t + i * 256;  // 4096 pairs
      int r = idx >> 5;
      int c2 = (idx & 31) << 1;
      *(unsigned*)&PQ[r * 72 + c2] = *(const unsigned*)&qsrc[r * 64 + c2];
    }
  }
  __syncthreads();
  v16h qf[2];
  qf[0] = load_a_frag(PQ, 72, wave * 16, 0, lane);
  qf[1] = load_a_frag(PQ, 72, wave * 16, 32, lane);
  __syncthreads();  // PQ now reused as per-wave P scratch

  _Float16* Pw = PQ + wave * 16 * 72;

  float Mrow[8], Lrow[8];
  v8f o[4];
#pragma unroll
  for (int j = 0; j < 8; ++j) { Mrow[j] = -1e30f; Lrow[j] = 0.0f; }
#pragma unroll
  for (int ct = 0; ct < 4; ++ct)
#pragma unroll
    for (int j = 0; j < 8; ++j) o[ct][j] = 0.0f;

  const float scale = 0.125f;  // 1/sqrt(64)

  for (int kb = 0; kb < Nn; kb += 64) {
    __syncthreads();
#if HAVE_TDM
    if (wave == 0)  // async DMA of K block (row-major, LDS row stride 144B)
      tdm_load_64x64_f16(kg + headbase + (size_t)kb * Dn,
                         (unsigned)(uintptr_t)&Ks[0], (unsigned)Nn);
#else
#pragma unroll
    for (int i = 0; i < 8; ++i) {  // K block 64x64, b32 copies
      int idx = t + i * 256;
      int r = idx >> 5;
      int c2 = (idx & 31) << 1;
      *(unsigned*)&Ks[r * 72 + c2] =
          *(const unsigned*)&kg[headbase + (size_t)(kb + r) * Dn + c2];
    }
#endif
    // V block transposed [d][key], 2 keys packed per b32 store
#pragma unroll
    for (int i = 0; i < 8; ++i) {
      int idx = t + i * 256;
      int dd = idx & 63;
      int kp2 = (idx >> 6) << 1;
      union { _Float16 hh[2]; unsigned u; } pk;
      pk.hh[0] = vg[headbase + (size_t)(kb + kp2) * Dn + dd];
      pk.hh[1] = vg[headbase + (size_t)(kb + kp2 + 1) * Dn + dd];
      *(unsigned*)&Vt[dd * 72 + kp2] = pk.u;
    }
#if HAVE_TDM
    if (wave == 0) __builtin_amdgcn_s_wait_tensorcnt(0);
#endif
    __syncthreads();

    // S strip (16 q x 64 keys) = Q . K^T   (B operand = Ks rows, contiguous)
    v8f s[4];
#pragma unroll
    for (int ct = 0; ct < 4; ++ct) {
      v8f sa = {};
#pragma unroll
      for (int ks = 0; ks < 2; ++ks) {
        v16h kf = load_bt_frag(Ks, 72, ks * 32, ct * 16, lane);
        sa = WMMA_F16(qf[ks], kf, sa);
      }
#pragma unroll
      for (int j = 0; j < 8; ++j) sa[j] *= scale;
      s[ct] = sa;
    }

    // online softmax: row max across 4 tiles + 16-lane halves
    float mnew[8];
#pragma unroll
    for (int j = 0; j < 8; ++j) mnew[j] = Mrow[j];
#pragma unroll
    for (int ct = 0; ct < 4; ++ct)
#pragma unroll
      for (int j = 0; j < 8; ++j) mnew[j] = fmaxf(mnew[j], s[ct][j]);
#pragma unroll
    for (int off = 8; off >= 1; off >>= 1)
#pragma unroll
      for (int j = 0; j < 8; ++j) mnew[j] = fmaxf(mnew[j], __shfl_xor(mnew[j], off, 32));

    float alpha[8], rs[8];
#pragma unroll
    for (int j = 0; j < 8; ++j) {
      alpha[j] = __expf(Mrow[j] - mnew[j]);
      Mrow[j] = mnew[j];
      rs[j] = 0.0f;
    }

    // P = exp(s - m); stash to per-wave LDS (C-layout -> A-layout transpose)
#pragma unroll
    for (int ct = 0; ct < 4; ++ct)
#pragma unroll
      for (int j = 0; j < 8; ++j) {
        float p = __expf(s[ct][j] - mnew[j]);
        rs[j] += p;
        Pw[(j + ((lane >> 4) << 3)) * 72 + ct * 16 + (lane & 15)] = (_Float16)p;
      }
#pragma unroll
    for (int off = 8; off >= 1; off >>= 1)
#pragma unroll
      for (int j = 0; j < 8; ++j) rs[j] += __shfl_xor(rs[j], off, 32);
#pragma unroll
    for (int j = 0; j < 8; ++j) Lrow[j] = Lrow[j] * alpha[j] + rs[j];
#pragma unroll
    for (int ct = 0; ct < 4; ++ct)
#pragma unroll
      for (int j = 0; j < 8; ++j) o[ct][j] *= alpha[j];

    // O += P (16x64) . V (64x64)   (B operand = Vt columns, contiguous)
#pragma unroll
    for (int ks = 0; ks < 2; ++ks) {
      v16h pf = load_a_frag(Pw, 72, 0, ks * 32, lane);
#pragma unroll
      for (int ct = 0; ct < 4; ++ct) {
        v16h vf = load_bt_frag(Vt, 72, ks * 32, ct * 16, lane);
        o[ct] = WMMA_F16(pf, vf, o[ct]);
      }
    }
  }

  // normalize + write [B, N, C] f16
#pragma unroll
  for (int ct = 0; ct < 4; ++ct) {
    int col = h * Dn + ct * 16 + (lane & 15);
#pragma unroll
    for (int j = 0; j < 8; ++j) {
      int nrow = qt * 128 + wave * 16 + j + ((lane >> 4) << 3);
      float inv = 1.0f / Lrow[j];
      og[((size_t)b * Nn + nrow) * Cn + col] = (_Float16)(o[ct][j] * inv);
    }
  }
}

// ---------------------------------------------------------------------------
extern "C" void kernel_launch(void* const* d_in, const int* in_sizes, int n_in,
                              void* d_out, int out_size, void* d_ws, size_t ws_size,
                              hipStream_t stream) {
  const float* x      = (const float*)d_in[0];
  const float* w_qkv  = (const float*)d_in[1];
  const float* w_proj = (const float*)d_in[2];
  const float* b_proj = (const float*)d_in[3];
  float* out = (float*)d_out;

  _Float16* qkv_h  = (_Float16*)d_ws;                   // [B*N, 3C]  24 MB
  _Float16* q_h    = qkv_h + (size_t)Bn * Nn * 3 * Cn;  // [B,H,N,d]   8 MB each
  _Float16* k_h    = q_h + (size_t)Bn * Hn * Nn * Dn;
  _Float16* v_h    = k_h + (size_t)Bn * Hn * Nn * Dn;
  _Float16* attn_h = v_h + (size_t)Bn * Hn * Nn * Dn;   // [B*N, C]    8 MB

  // 1) qkv = x @ w_qkv  (f32 in, f16 out)
  gemm_wmma_kernel<false, true, false>
      <<<dim3((3 * Cn) / 128, (Bn * Nn) / 128), 256, 0, stream>>>(
          x, w_qkv, nullptr, qkv_h, Bn * Nn, 3 * Cn, Cn);

  // 2) RoPE + pack to [B,H,N,d]
  rope_pack_kernel<<<(Bn * Hn * Nn * 32) / 256, 256, 0, stream>>>(qkv_h, q_h, k_h, v_h);

  // 3) flash attention
  flash_attn_kernel<<<Bn * Hn * (Nn / 128), 256, 0, stream>>>(q_h, k_h, v_h, attn_h);

  // 4) out = attn @ w_proj + b_proj  (f16 in, f32 out)
  gemm_wmma_kernel<true, false, true>
      <<<dim3(Cn / 128, (Bn * Nn) / 128), 256, 0, stream>>>(
          attn_h, w_proj, b_proj, out, Bn * Nn, Cn, Cn);
}
